// SelfAttention_69965017251966
// MI455X (gfx1250) — matrix operand order repaired and checked by reading, not compiled, not run on hardware
//
#include <hip/hip_runtime.h>

// ---------------------------------------------------------------------------
// Types for CDNA5 WMMA (wave32): v16bf A/B fragments, v8f C/D accumulator
// ---------------------------------------------------------------------------
typedef __attribute__((ext_vector_type(16))) __bf16 v16bf;
typedef __attribute__((ext_vector_type(8)))  float  v8f;

union Frag {
    uint4 u4[2];   // 32 bytes
    v16bf bf;
};

__device__ __forceinline__ unsigned short f2bf(float x) {
    union { float f; unsigned u; } v; v.f = x;
    unsigned r = v.u + 0x7FFFu + ((v.u >> 16) & 1u);   // round-to-nearest-even
    return (unsigned short)(r >> 16);
}

__device__ __forceinline__ v8f wmma_bf16(const Frag& a, const Frag& b, v8f c) {
    return __builtin_amdgcn_wmma_f32_16x16x32_bf16(
        false, a.bf, false, b.bf, (short)0, c, false, false);
}

// LDS byte address of a __shared__ object = low 32 bits of its flat address
// (CDNA5 aperture rule).
__device__ __forceinline__ unsigned lds_addr(const void* p) {
    return (unsigned)(unsigned long long)(size_t)p;
}

// Async copy 32 B / lane: memory -> LDS (GLOBAL_LOAD_ASYNC_TO_LDS_B128,
// ASYNCcnt += 2; INST_OFFSET applies to both LDS and memory addresses).
__device__ __forceinline__ void async_copy_32B(unsigned lds, const void* g) {
    unsigned long long ga = (unsigned long long)(size_t)g;
    asm volatile(
        "global_load_async_to_lds_b128 %0, %1, off\n\t"
        "global_load_async_to_lds_b128 %0, %1, off offset:16"
        :: "v"(lds), "v"(ga) : "memory");
}

// Async copy 64 B / lane (4 x b128)
__device__ __forceinline__ void async_copy_64B(unsigned lds, const void* g) {
    unsigned long long ga = (unsigned long long)(size_t)g;
    asm volatile(
        "global_load_async_to_lds_b128 %0, %1, off\n\t"
        "global_load_async_to_lds_b128 %0, %1, off offset:16\n\t"
        "global_load_async_to_lds_b128 %0, %1, off offset:32\n\t"
        "global_load_async_to_lds_b128 %0, %1, off offset:48"
        :: "v"(lds), "v"(ga) : "memory");
}

#define WAIT_ASYNC_4() asm volatile("s_wait_asynccnt 0x4" ::: "memory")

// ---------------------------------------------------------------------------
// f32 -> bf16 conversion (grid-stride)
// ---------------------------------------------------------------------------
__global__ __launch_bounds__(256) void cvt_bf16(const float* __restrict__ in,
                                                unsigned short* __restrict__ out,
                                                int n) {
    for (int i = blockIdx.x * blockDim.x + threadIdx.x; i < n;
         i += gridDim.x * blockDim.x)
        out[i] = f2bf(in[i]);
}

// ---------------------------------------------------------------------------
// f32 [K,N] -> bf16 transposed [N,K], tiled 32x32 through LDS
// ---------------------------------------------------------------------------
__global__ __launch_bounds__(256) void cvt_transpose_bf16(
    const float* __restrict__ W,        // [K,N] f32
    unsigned short* __restrict__ Wt,    // [N,K] bf16
    int K, int N)
{
    __shared__ unsigned short tile[32][36];
    const int t  = threadIdx.x;
    const int n0 = blockIdx.x * 32;
    const int k0 = blockIdx.y * 32;
    {
        int kk = t >> 3, n4 = (t & 7) * 4;
        float4 v = *(const float4*)(W + (size_t)(k0 + kk) * N + n0 + n4);
        tile[kk][n4 + 0] = f2bf(v.x);
        tile[kk][n4 + 1] = f2bf(v.y);
        tile[kk][n4 + 2] = f2bf(v.z);
        tile[kk][n4 + 3] = f2bf(v.w);
    }
    __syncthreads();
    {
        int nn = t >> 3, k4 = (t & 7) * 4;
        unsigned a0 = tile[k4 + 0][nn], a1 = tile[k4 + 1][nn];
        unsigned a2 = tile[k4 + 2][nn], a3 = tile[k4 + 3][nn];
        uint2 o;
        o.x = a0 | (a1 << 16);
        o.y = a2 | (a3 << 16);
        *(uint2*)(Wt + (size_t)(n0 + nn) * K + k0 + k4) = o;
    }
}

// ---------------------------------------------------------------------------
// V slice of qkv [B*S, 3D] -> vT [B*H, HD, S] (bf16 transpose, tiled 32x32)
// ---------------------------------------------------------------------------
__global__ __launch_bounds__(256) void transpose_v(
    const unsigned short* __restrict__ qkv,  // [B*S, 3D] bf16
    unsigned short* __restrict__ vT,         // [B*H, 64, S] bf16
    int S, int D)
{
    __shared__ unsigned short tile[32][36];
    const int t  = threadIdx.x;
    const int s0 = blockIdx.x * 32;
    const int c0 = blockIdx.y * 32;          // hd sub-tile (0 or 32)
    const int bh = blockIdx.z;               // b*H + h
    const int b  = bh >> 4, h = bh & 15;
    const int D3 = 3 * D;
    {
        int ss = t >> 3, c4 = (t & 7) * 4;
        const unsigned short* src =
            qkv + (size_t)(b * S + s0 + ss) * D3 + 2 * D + h * 64 + c0 + c4;
        uint2 v = *(const uint2*)src;
        const unsigned short* p = (const unsigned short*)&v;
        tile[ss][c4 + 0] = p[0];
        tile[ss][c4 + 1] = p[1];
        tile[ss][c4 + 2] = p[2];
        tile[ss][c4 + 3] = p[3];
    }
    __syncthreads();
    {
        int cc = t >> 3, s4 = (t & 7) * 4;
        unsigned a0 = tile[s4 + 0][cc], a1 = tile[s4 + 1][cc];
        unsigned a2 = tile[s4 + 2][cc], a3 = tile[s4 + 3][cc];
        uint2 o;
        o.x = a0 | (a1 << 16);
        o.y = a2 | (a3 << 16);
        *(uint2*)(vT + ((size_t)bh * 64 + c0 + cc) * S + s0 + s4) = o;
    }
}

// ---------------------------------------------------------------------------
// Tiled bf16 GEMM: C[M,N] = A[M,K] * Bt[N,K]^T + bias, out bf16 or f32.
// Block = 256 threads = 8 waves; block tile 128x128; waves 4x2, wave tile
// 32x64 (8 WMMA / K-step). Double-buffered LDS filled by async-to-LDS copies.
// Branchless pipeline: next tile always issued (clamped to last), so the
// hot loop has no control flow; s_wait_asynccnt 4 retires the current tile.
// ---------------------------------------------------------------------------
#define GT_M 128
#define GT_N 128
#define GT_K 32
#define ASTR 40
#define BSTR 40

__global__ __launch_bounds__(256) void gemm_bf16(
    const unsigned short* __restrict__ A,    // [M,K] bf16
    const unsigned short* __restrict__ Bt,   // [N,K] bf16 (pre-transposed)
    const float* __restrict__ bias,          // [N]
    unsigned short* __restrict__ Cbf,        // bf16 out (or null)
    float* __restrict__ Cf,                  // f32 out  (or null)
    int M, int N, int K)
{
    __shared__ unsigned short As[2][GT_M * ASTR];   // 2 x 10240 B (m x k)
    __shared__ unsigned short Bs[2][GT_N * BSTR];   // 2 x 10240 B (n x k)

    const int t    = threadIdx.x;
    const int lane = t & 31;
    const int wave = t >> 5;          // 0..7
    const int wm   = wave >> 1;       // 0..3  (M dir, 32 rows each)
    const int wn   = wave & 1;        // 0..1  (N dir, 64 cols each)
    const int lh   = lane >> 4;       // lane half
    const int ll   = lane & 15;
    const int m0   = blockIdx.x * GT_M;
    const int n0   = blockIdx.y * GT_N;

    // staging assignment: 128 rows x 32 k, 16 bf16 (32 B) per thread
    const int srow = t >> 1, scol = (t & 1) * 16;
    const int sidx = srow * ASTR + scol;
    const unsigned short* Ag = A  + (size_t)(m0 + srow) * K + scol;
    const unsigned short* Bg = Bt + (size_t)(n0 + srow) * K + scol;

    v8f c[2][4] = {};

    const int nk = K / GT_K;

    // prologue: tile 0 -> buffer 0
    async_copy_32B(lds_addr(&As[0][sidx]), Ag);
    async_copy_32B(lds_addr(&Bs[0][sidx]), Bg);

    for (int kt = 0; kt < nk; ++kt) {
        __syncthreads();   // all waves done reading buffer (kt+1)&1 (iter kt-1)

        // always issue next tile (clamped; final-iter duplicate is harmless:
        // it targets the buffer with no remaining readers)
        const int ktn  = (kt + 1 < nk) ? kt + 1 : nk - 1;
        const int bufn = (kt + 1) & 1;
        async_copy_32B(lds_addr(&As[bufn][sidx]), Ag + ktn * GT_K);
        async_copy_32B(lds_addr(&Bs[bufn][sidx]), Bg + ktn * GT_K);
        WAIT_ASYNC_4();    // in-order completion => tile kt landed
        __syncthreads();   // every wave's tile-kt data visible

        // speculative L2 warm-up (safe past end: dropped on xlat failure)
        __builtin_prefetch(Ag + (kt + 2) * GT_K, 0, 1);
        __builtin_prefetch(Bg + (kt + 2) * GT_K, 0, 1);

        const unsigned short* as = As[kt & 1];
        const unsigned short* bs = Bs[kt & 1];

        // ---- fragments (ISA 16-bit A 16x32 / B 32x16 lane layouts) ----
        Frag a[2], b[4];
#pragma unroll
        for (int mi = 0; mi < 2; ++mi) {
            const unsigned short* p =
                &as[(wm * 32 + mi * 16 + ll) * ASTR + lh * 8];
            a[mi].u4[0] = *(const uint4*)(p);        // K = lh*8 + 0..7
            a[mi].u4[1] = *(const uint4*)(p + 16);   // K = 16 + lh*8 + 0..7
        }
#pragma unroll
        for (int ni = 0; ni < 4; ++ni) {
            const unsigned short* p =
                &bs[(wn * 64 + ni * 16 + ll) * BSTR + lh * 8];
            b[ni].u4[0] = *(const uint4*)(p);        // K = lh*8 + 0..7
            b[ni].u4[1] = *(const uint4*)(p + 16);   // K = 16 + lh*8 + 0..7
        }
#pragma unroll
        for (int mi = 0; mi < 2; ++mi)
#pragma unroll
            for (int ni = 0; ni < 4; ++ni)
                c[mi][ni] = wmma_bf16(a[mi], b[ni], c[mi][ni]);
    }

    // ---- epilogue: C layout row = elem + 8*(lane/16), col = lane%16 ----
#pragma unroll
    for (int mi = 0; mi < 2; ++mi)
#pragma unroll
        for (int ni = 0; ni < 4; ++ni)
#pragma unroll
            for (int r = 0; r < 8; ++r) {
                int row = m0 + wm * 32 + mi * 16 + r + 8 * lh;
                int col = n0 + wn * 64 + ni * 16 + ll;
                float v = c[mi][ni][r] + bias[col];
                if (Cbf) Cbf[(size_t)row * N + col] = f2bf(v);
                else     Cf [(size_t)row * N + col] = v;
            }
}

// ---------------------------------------------------------------------------
// Flash attention: one block per (q-tile of 128, head, batch).
// 8 waves x 16 query rows; 64-key blocks; online softmax in f32.
// Q/K/V staged by async-to-LDS copies; K/V double-buffered so block j+1's
// copies overlap block j's WMMAs + softmax (branchless clamped issue).
// ---------------------------------------------------------------------------
#define FA_QT 128
#define FA_KT 64
#define FA_HD 64
#define QSTR 72
#define KSTR 72
#define VSTR 72
#define PSTR 72

__global__ __launch_bounds__(256) void flash_attn(
    const unsigned short* __restrict__ qkv,  // [B*S, 3*D] bf16
    const unsigned short* __restrict__ vT,   // [B*H, 64, S] bf16
    unsigned short* __restrict__ outp,       // [B*S, D]   bf16
    int S, int D)
{
    __shared__ unsigned short Qs[FA_QT * QSTR];        // 18432 B
    __shared__ unsigned short Ks[2][FA_KT * KSTR];     // 2 x 9216 B (key x hd)
    __shared__ unsigned short Vt[2][FA_HD * VSTR];     // 2 x 9216 B (hd x key)
    __shared__ unsigned short Ps[8 * 16 * PSTR];       // 18432 B (per-wave P)

    const int t    = threadIdx.x;
    const int lane = t & 31;
    const int wave = t >> 5;
    const int lh   = lane >> 4;
    const int ll   = lane & 15;
    const int qt   = blockIdx.x;
    const int h    = blockIdx.y;
    const int b    = blockIdx.z;
    const int bh   = b * 16 + h;
    const int D3   = 3 * D;
    const size_t rowBase = (size_t)b * S;
    const int qcol = h * FA_HD;
    const int kcol = D + h * FA_HD;
    const float scale = 0.125f;              // 1/sqrt(64)

    // staging assignments
    const int qr = t >> 1, qc = (t & 1) * 32;     // Q: 128 x 64, 64 B/thread
    const int kr = t >> 2, kc = (t & 3) * 16;     // K/V: 64 x 64, 32 B/thread
    const int kidx = kr * KSTR + kc;
    const unsigned short* Kg = qkv + (rowBase + kr) * D3 + kcol + kc;   // + kb*D3
    const unsigned short* Vg = vT + ((size_t)bh * 64 + kr) * S + kc;    // + kb

    // ---- stage Q tile + K/V block 0 (async) ----
    async_copy_64B(lds_addr(&Qs[qr * QSTR + qc]),
                   qkv + (rowBase + qt * FA_QT + qr) * D3 + qcol + qc);
    async_copy_32B(lds_addr(&Ks[0][kidx]), Kg);
    async_copy_32B(lds_addr(&Vt[0][kidx]), Vg);
    WAIT_ASYNC_4();    // Q's 4 copies retired; K/V block 0 still in flight
    __syncthreads();

    // ---- Q fragments are invariant across key blocks: hoist ----
    Frag aq[2];
#pragma unroll
    for (int kc2 = 0; kc2 < 2; ++kc2) {
        const unsigned short* p =
            &Qs[(wave * 16 + ll) * QSTR + kc2 * 32 + lh * 8];
        aq[kc2].u4[0] = *(const uint4*)(p);
        aq[kc2].u4[1] = *(const uint4*)(p + 16);
    }

    v8f o[4] = {};
    float mrow[8], lrow[8];
#pragma unroll
    for (int r = 0; r < 8; ++r) { mrow[r] = -1e30f; lrow[r] = 0.0f; }

    const int nb = S / FA_KT;
    for (int j = 0; j < nb; ++j) {
        __syncthreads();   // all waves done reading buffer (j+1)&1 (iter j-1)

        // issue next K/V block (clamped on the last iteration)
        const int kbn  = (j + 1 < nb) ? (j + 1) * FA_KT : j * FA_KT;
        const int bufn = (j + 1) & 1;
        async_copy_32B(lds_addr(&Ks[bufn][kidx]), Kg + (size_t)kbn * D3);
        async_copy_32B(lds_addr(&Vt[bufn][kidx]), Vg + kbn);
        WAIT_ASYNC_4();    // block j's copies landed
        __syncthreads();

        const unsigned short* ksb = Ks[j & 1];
        const unsigned short* vtb = Vt[j & 1];

        // ---- S = scale * Q K^T  (wave's 16 rows x 64 keys) ----
        v8f sc[4];
#pragma unroll
        for (int nt = 0; nt < 4; ++nt) {
            v8f acc = {};
#pragma unroll
            for (int kc2 = 0; kc2 < 2; ++kc2) {
                Frag bk;
                const unsigned short* p =
                    &ksb[(nt * 16 + ll) * KSTR + kc2 * 32 + lh * 8];
                bk.u4[0] = *(const uint4*)(p);
                bk.u4[1] = *(const uint4*)(p + 16);
                acc = wmma_bf16(aq[kc2], bk, acc);
            }
#pragma unroll
            for (int r = 0; r < 8; ++r) acc[r] *= scale;
            sc[nt] = acc;
        }

        // ---- online softmax (row = elem + 8*lane_half; reduce over 16 lanes) ----
        float mnew[8], corr[8];
#pragma unroll
        for (int r = 0; r < 8; ++r) {
            float mx = sc[0][r];
#pragma unroll
            for (int nt = 1; nt < 4; ++nt) mx = fmaxf(mx, sc[nt][r]);
#pragma unroll
            for (int off = 1; off < 16; off <<= 1)
                mx = fmaxf(mx, __shfl_xor(mx, off, 32));
            mnew[r] = fmaxf(mrow[r], mx);
            corr[r] = __expf(mrow[r] - mnew[r]);
        }
#pragma unroll
        for (int nt = 0; nt < 4; ++nt)
#pragma unroll
            for (int r = 0; r < 8; ++r)
                sc[nt][r] = __expf(sc[nt][r] - mnew[r]);
#pragma unroll
        for (int r = 0; r < 8; ++r) {
            float s = 0.0f;
#pragma unroll
            for (int nt = 0; nt < 4; ++nt) s += sc[nt][r];
#pragma unroll
            for (int off = 1; off < 16; off <<= 1)
                s += __shfl_xor(s, off, 32);
            lrow[r] = lrow[r] * corr[r] + s;
            mrow[r] = mnew[r];
        }
#pragma unroll
        for (int ht = 0; ht < 4; ++ht)
#pragma unroll
            for (int r = 0; r < 8; ++r) o[ht][r] *= corr[r];

        // ---- P: C-layout -> LDS -> A-layout fragments ----
#pragma unroll
        for (int nt = 0; nt < 4; ++nt)
#pragma unroll
            for (int r = 0; r < 8; ++r)
                Ps[(wave * 16 + r + 8 * lh) * PSTR + nt * 16 + ll] =
                    f2bf(sc[nt][r]);
        __syncthreads();

        // ---- O += P V ----
        Frag ap[2];
#pragma unroll
        for (int kc2 = 0; kc2 < 2; ++kc2) {
            const unsigned short* p =
                &Ps[(wave * 16 + ll) * PSTR + kc2 * 32 + lh * 8];
            ap[kc2].u4[0] = *(const uint4*)(p);
            ap[kc2].u4[1] = *(const uint4*)(p + 16);
        }
#pragma unroll
        for (int ht = 0; ht < 4; ++ht) {
#pragma unroll
            for (int kc2 = 0; kc2 < 2; ++kc2) {
                Frag bv;
                const unsigned short* p =
                    &vtb[(ht * 16 + ll) * VSTR + kc2 * 32 + lh * 8];
                bv.u4[0] = *(const uint4*)(p);
                bv.u4[1] = *(const uint4*)(p + 16);
                o[ht] = wmma_bf16(ap[kc2], bv, o[ht]);
            }
        }
    }

    // ---- write normalized output tile (bf16, [B*S, D] at head column) ----
#pragma unroll
    for (int ht = 0; ht < 4; ++ht)
#pragma unroll
        for (int r = 0; r < 8; ++r) {
            int row = qt * FA_QT + wave * 16 + r + 8 * lh;
            int col = h * FA_HD + ht * 16 + ll;
            float v = o[ht][r] / lrow[r];
            outp[(rowBase + row) * (size_t)D + col] = f2bf(v);
        }
}

// ---------------------------------------------------------------------------
// Launch: cvt -> GEMM(qkv) -> transpose V -> flash-attn -> GEMM(proj)
// ---------------------------------------------------------------------------
extern "C" void kernel_launch(void* const* d_in, const int* in_sizes, int n_in,
                              void* d_out, int out_size, void* d_ws, size_t ws_size,
                              hipStream_t stream) {
    const float* x     = (const float*)d_in[0];
    const float* Wqkv  = (const float*)d_in[1];
    const float* bqkv  = (const float*)d_in[2];
    const float* Wproj = (const float*)d_in[3];
    const float* bproj = (const float*)d_in[4];
    float* out = (float*)d_out;

    const int Bb = 2, S = 2048, D = 1024, H = 16;
    const int M  = Bb * S;      // 4096
    const int N1 = 3 * D;       // 3072

    char* ws = (char*)d_ws;
    unsigned short* x_bf      = (unsigned short*)(ws);               //  8 MiB
    unsigned short* wqkvT_bf  = (unsigned short*)(ws + 8388608);     //  6 MiB [3072,1024]
    unsigned short* wprojT_bf = (unsigned short*)(ws + 14680064);    //  2 MiB [1024,1024]
    unsigned short* qkv_bf    = (unsigned short*)(ws + 16777216);    // 24 MiB
    unsigned short* attn_bf   = (unsigned short*)(ws + 41943040);    //  8 MiB
    unsigned short* vT_bf     = (unsigned short*)(ws + 50331648);    //  8 MiB [B*H,64,S]

    cvt_bf16<<<1024, 256, 0, stream>>>(x, x_bf, M * D);

    dim3 gt1(N1 / 32, D / 32);   // W_qkv [D, 3D] -> [3D, D]
    cvt_transpose_bf16<<<gt1, 256, 0, stream>>>(Wqkv, wqkvT_bf, D, N1);
    dim3 gt2(D / 32, D / 32);    // W_proj [D, D] -> [D, D]^T
    cvt_transpose_bf16<<<gt2, 256, 0, stream>>>(Wproj, wprojT_bf, D, D);

    dim3 g1(M / GT_M, N1 / GT_N);
    gemm_bf16<<<g1, 256, 0, stream>>>(x_bf, wqkvT_bf, bqkv, qkv_bf, nullptr,
                                      M, N1, D);

    dim3 gv(S / 32, 2, Bb * H);
    transpose_v<<<gv, 256, 0, stream>>>(qkv_bf, vT_bf, S, D);

    dim3 ga(S / FA_QT, H, Bb);
    flash_attn<<<ga, 256, 0, stream>>>(qkv_bf, vT_bf, attn_bf, S, D);

    dim3 g2(M / GT_M, D / GT_N);
    gemm_bf16<<<g2, 256, 0, stream>>>(attn_bf, wprojT_bf, bproj, nullptr, out,
                                      M, D, D);
}
